// GENI_1666447311032
// MI455X (gfx1250) — compile-verified, matrix operand order checked by reading, not run on hardware
//
#include <hip/hip_runtime.h>
#include <hip/hip_bf16.h>

typedef __attribute__((ext_vector_type(16))) _Float16 v16h;
typedef __attribute__((ext_vector_type(8)))  _Float16 v8h;
typedef __attribute__((ext_vector_type(8)))  float    v8f;
typedef __attribute__((ext_vector_type(4)))  float    v4f;

#define H_HEADS 4
#define IN_DIM  128
#define HID     64
#define COLS    256      // H_HEADS * HID
#define PRED    16
#define RELS    50
#define LAYERS  3

// ---- ordered-int mapping so atomicMax(int) implements float max ----
__device__ __forceinline__ int f2key(float f) {
    int i = __float_as_int(f);
    return (i >= 0) ? i : (i ^ 0x7FFFFFFF);
}
__device__ __forceinline__ float key2f(int i) {
    return __int_as_float((i >= 0) ? i : (i ^ 0x7FFFFFFF));
}

// ---------------- W1 (H,hid,IN_DIM) fp32 -> contiguous f16 [col][k] --------
// flat index is identical: col = head*64 + j, element = col*128 + k
__global__ void k_w1_to_f16(const float* __restrict__ W1,
                            _Float16* __restrict__ W1h, int n) {
    int i = blockIdx.x * blockDim.x + threadIdx.x;
    if (i < n) W1h[i] = (_Float16)W1[i];
}

// ---------------- predicate attention table: etab[l][r][h] -----------------
__global__ void k_etab(const float* __restrict__ rel_emb,
                       const float* __restrict__ W_pred,
                       float* __restrict__ etab) {
    int i = blockIdx.x * blockDim.x + threadIdx.x;
    if (i >= LAYERS * RELS * H_HEADS) return;
    int l = i / (RELS * H_HEADS);
    int rem = i % (RELS * H_HEADS);
    int r = rem / H_HEADS;
    int h = rem % H_HEADS;
    float e = 0.f;
    #pragma unroll
    for (int p = 0; p < PRED; ++p)
        e += rel_emb[r * PRED + p] * W_pred[(l * PRED + p) * H_HEADS + h];
    etab[i] = (e > 0.f) ? e : 0.2f * e;   // leaky_relu slope 0.2
}

// ---------------- WMMA node-scoring MLP: h[n,head] -------------------------
// One wave32 handles a 16-node tile. Per head: 4 accum fragments over 64 cols,
// K=128 consumed in 4 chunks of 32 via v_wmma_f32_16x16x32_f16.
__global__ void k_mlp_wmma(const float* __restrict__ inputs,
                           const _Float16* __restrict__ W1h,
                           const float* __restrict__ b1,
                           const float* __restrict__ W2,
                           const float* __restrict__ b2,
                           float* __restrict__ hout,
                           int n, int nTiles) {
    const int wave = (blockIdx.x * blockDim.x + threadIdx.x) >> 5;
    if (wave >= nTiles) return;
    const int lane = threadIdx.x & 31;
    const int grp  = lane >> 4;      // half-group: K/row-group selector
    const int l15  = lane & 15;
    const int nodeBase = wave * 16;
    // A fragment source row for this lane (clamped for a possible tail tile)
    int arow = nodeBase + l15;
    if (arow > n - 1) arow = n - 1;
    const float* __restrict__ abase = inputs + (size_t)arow * IN_DIM;

    const v8f zero = {};

    #pragma unroll
    for (int head = 0; head < H_HEADS; ++head) {
        v8f acc0 = zero, acc1 = zero, acc2 = zero, acc3 = zero;

        #pragma unroll
        for (int kk = 0; kk < IN_DIM; kk += 32) {
            const int kb = kk + grp * 8;
            // ---- A fragment: f32 -> f16, layout per ISA 16-bit A table ----
            v4f a0 = *(const v4f*)(abase + kb + 0);
            v4f a1 = *(const v4f*)(abase + kb + 4);
            v4f a2 = *(const v4f*)(abase + kb + 16);
            v4f a3 = *(const v4f*)(abase + kb + 20);
            v16h A;
            #pragma unroll
            for (int i = 0; i < 4; ++i) {
                A[i]      = (_Float16)a0[i];
                A[4 + i]  = (_Float16)a1[i];
                A[8 + i]  = (_Float16)a2[i];
                A[12 + i] = (_Float16)a3[i];
            }
            // ---- B fragments for the 4 column tiles of this head ----------
            #pragma unroll
            for (int tt = 0; tt < 4; ++tt) {
                const int col = head * HID + tt * 16 + l15;
                const _Float16* q = W1h + (size_t)col * IN_DIM + kb;
                v8h blo = *(const v8h*)(q);
                v8h bhi = *(const v8h*)(q + 16);
                v16h B;
                #pragma unroll
                for (int i = 0; i < 8; ++i) { B[i] = blo[i]; B[8 + i] = bhi[i]; }
                v8f c = (tt == 0) ? acc0 : (tt == 1) ? acc1 : (tt == 2) ? acc2 : acc3;
                c = __builtin_amdgcn_wmma_f32_16x16x32_f16(
                        false, A, false, B, (short)0, c, false, false);
                if (tt == 0) acc0 = c; else if (tt == 1) acc1 = c;
                else if (tt == 2) acc2 = c; else acc3 = c;
            }
        }

        // ---- fused epilogue: relu(acc + b1) * W2, reduce across 64 cols ----
        v8f hacc = zero;
        #pragma unroll
        for (int tt = 0; tt < 4; ++tt) {
            const int col  = head * HID + tt * 16 + l15;
            const float bias = b1[col];
            const float w2   = W2[col];
            v8f c = (tt == 0) ? acc0 : (tt == 1) ? acc1 : (tt == 2) ? acc2 : acc3;
            #pragma unroll
            for (int v = 0; v < 8; ++v) {
                float p = c[v] + bias;
                p = (p > 0.f) ? p : 0.f;
                p *= w2;
                // butterfly sum across the 16-lane half-group (cols of tile)
                p += __shfl_xor(p, 1, 32);
                p += __shfl_xor(p, 2, 32);
                p += __shfl_xor(p, 4, 32);
                p += __shfl_xor(p, 8, 32);
                hacc[v] += p;
            }
        }
        if (l15 == 0) {
            #pragma unroll
            for (int v = 0; v < 8; ++v) {
                const int node = nodeBase + v + 8 * grp;   // C/D row layout
                if (node < n)
                    hout[node * H_HEADS + head] = hacc[v] + b2[head];
            }
        }
    }
}

// ---------------- per-layer scratch init -----------------------------------
__global__ void k_init_nh(int* __restrict__ m, float* __restrict__ s,
                          float* __restrict__ u, int count) {
    int i = blockIdx.x * blockDim.x + threadIdx.x;
    if (i < count) { m[i] = (int)0x80000000; s[i] = 0.f; u[i] = 0.f; }
}

// ---------------- segment max over incoming edges ---------------------------
__global__ void k_edge_max(const int* __restrict__ et, const int* __restrict__ dst,
                           const float* __restrict__ etab_l,
                           int* __restrict__ m, int E) {
    int i = blockIdx.x * blockDim.x + threadIdx.x;
    if (i >= E) return;
    __builtin_prefetch(et + i + 8192, 0, 0);
    __builtin_prefetch(dst + i + 8192, 0, 0);
    const int t = et[i];
    const int d = dst[i];
    const float* e = etab_l + t * H_HEADS;
    #pragma unroll
    for (int h = 0; h < H_HEADS; ++h)
        atomicMax(&m[d * H_HEADS + h], f2key(e[h]));
}

// ------- fused softmax-numerator + weighted aggregate (s, u) ----------------
__global__ void k_edge_agg(const int* __restrict__ et, const int* __restrict__ src,
                           const int* __restrict__ dst,
                           const float* __restrict__ etab_l,
                           const int* __restrict__ m,
                           const float* __restrict__ hcur,
                           float* __restrict__ s, float* __restrict__ u, int E) {
    int i = blockIdx.x * blockDim.x + threadIdx.x;
    if (i >= E) return;
    __builtin_prefetch(et + i + 8192, 0, 0);
    __builtin_prefetch(src + i + 8192, 0, 0);
    __builtin_prefetch(dst + i + 8192, 0, 0);
    const int t  = et[i];
    const int sn = src[i];
    const int d  = dst[i];
    const float* e = etab_l + t * H_HEADS;
    #pragma unroll
    for (int h = 0; h < H_HEADS; ++h) {
        const float mv = key2f(m[d * H_HEADS + h]);
        const float w  = __expf(e[h] - mv);
        atomicAdd(&s[d * H_HEADS + h], w);
        atomicAdd(&u[d * H_HEADS + h], w * hcur[sn * H_HEADS + h]);
    }
}

// ---------------- node update: normalize, relu, optional mean-repeat --------
__global__ void k_node_update(const float* __restrict__ s, const float* __restrict__ u,
                              float* __restrict__ hnext, int n, int do_mean) {
    int i = blockIdx.x * blockDim.x + threadIdx.x;
    if (i >= n) return;
    float v[H_HEADS];
    #pragma unroll
    for (int h = 0; h < H_HEADS; ++h) {
        const float sv = s[i * H_HEADS + h];
        float x = (sv > 0.f) ? (u[i * H_HEADS + h] / sv) : 0.f;  // no edges -> 0
        v[h] = (x > 0.f) ? x : 0.f;                               // relu
    }
    if (do_mean) {
        const float mn = 0.25f * (v[0] + v[1] + v[2] + v[3]);
        #pragma unroll
        for (int h = 0; h < H_HEADS; ++h) v[h] = mn;
    }
    #pragma unroll
    for (int h = 0; h < H_HEADS; ++h) hnext[i * H_HEADS + h] = v[h];
}

// ---------------- centrality scaling + final leaky_relu(mean) ---------------
__global__ void k_final(const float* __restrict__ hfin,
                        const float* __restrict__ centrality,
                        const float* __restrict__ gamma,
                        const float* __restrict__ beta,
                        float* __restrict__ out, int n) {
    int i = blockIdx.x * blockDim.x + threadIdx.x;
    if (i >= n) return;
    const float c = centrality[i];
    float acc = 0.f;
    #pragma unroll
    for (int h = 0; h < H_HEADS; ++h)
        acc += (c * gamma[h] + beta[h]) * hfin[i * H_HEADS + h];
    acc *= 0.25f;
    out[i] = (acc > 0.f) ? acc : 0.01f * acc;
}

extern "C" void kernel_launch(void* const* d_in, const int* in_sizes, int n_in,
                              void* d_out, int out_size, void* d_ws, size_t ws_size,
                              hipStream_t stream) {
    const float* inputs     = (const float*)d_in[0];
    const float* centrality = (const float*)d_in[1];
    const float* W1         = (const float*)d_in[2];
    const float* b1         = (const float*)d_in[3];
    const float* W2         = (const float*)d_in[4];
    const float* b2         = (const float*)d_in[5];
    const float* rel_emb    = (const float*)d_in[6];
    const float* W_pred     = (const float*)d_in[7];
    const float* gamma      = (const float*)d_in[8];
    const float* beta       = (const float*)d_in[9];
    const int*   edge_types = (const int*)d_in[10];
    const int*   src        = (const int*)d_in[11];
    const int*   dst        = (const int*)d_in[12];
    float* out = (float*)d_out;

    const int N = in_sizes[1];    // centrality has N elements
    const int E = in_sizes[10];   // edge_types has E elements

    // ---- workspace carve-up ----
    char* ws = (char*)d_ws;
    _Float16* W1h = (_Float16*)ws;                       // 256*128*2  = 64 KB
    float* etab   = (float*)(ws + 65536);                // 600 floats
    float* hA     = (float*)(ws + 69632);                // N*4 floats
    float* hB     = hA + (size_t)N * H_HEADS;
    int*   m      = (int*)(hB + (size_t)N * H_HEADS);
    float* s      = (float*)(m + (size_t)N * H_HEADS);
    float* u      = s + (size_t)N * H_HEADS;

    const int TPB = 256;

    // W1 -> f16 copy (layout is already [col][k])
    {
        int cnt = COLS * IN_DIM;
        k_w1_to_f16<<<(cnt + TPB - 1) / TPB, TPB, 0, stream>>>(W1, W1h, cnt);
    }
    // predicate attention table (tiny)
    {
        int cnt = LAYERS * RELS * H_HEADS;
        k_etab<<<(cnt + TPB - 1) / TPB, TPB, 0, stream>>>(rel_emb, W_pred, etab);
    }
    // WMMA scoring MLP -> hA
    {
        int nTiles = (N + 15) / 16;
        int wavesPerBlock = 4;                     // 128 threads
        int blocks = (nTiles + wavesPerBlock - 1) / wavesPerBlock;
        k_mlp_wmma<<<blocks, wavesPerBlock * 32, 0, stream>>>(
            inputs, W1h, b1, W2, b2, hA, N, nTiles);
    }

    float* hcur = hA;
    float* hnxt = hB;
    const int nhCount = N * H_HEADS;
    const int nhBlocks = (nhCount + TPB - 1) / TPB;
    const int eBlocks  = (E + TPB - 1) / TPB;
    const int nBlocks  = (N + TPB - 1) / TPB;

    for (int l = 0; l < LAYERS; ++l) {
        const float* etab_l = etab + l * RELS * H_HEADS;
        k_init_nh<<<nhBlocks, TPB, 0, stream>>>(m, s, u, nhCount);
        k_edge_max<<<eBlocks, TPB, 0, stream>>>(edge_types, dst, etab_l, m, E);
        k_edge_agg<<<eBlocks, TPB, 0, stream>>>(edge_types, src, dst, etab_l,
                                                m, hcur, s, u, E);
        k_node_update<<<nBlocks, TPB, 0, stream>>>(s, u, hnxt, N,
                                                   (l != LAYERS - 1) ? 1 : 0);
        float* tmp = hcur; hcur = hnxt; hnxt = tmp;
    }

    k_final<<<nBlocks, TPB, 0, stream>>>(hcur, centrality, gamma, beta, out, N);
}
